// HypergraphConstructor_62577673503459
// MI455X (gfx1250) — matrix operand order repaired and checked by reading, not compiled, not run on hardware
//
#include <hip/hip_runtime.h>
#include <hip/hip_fp16.h>

#define NNODES 16384
#define DIM    512
#define ALPHA  3.0f
#define TOPK   10

typedef __attribute__((ext_vector_type(16))) _Float16 v16h;
typedef __attribute__((ext_vector_type(4)))  _Float16 v4h;
typedef __attribute__((ext_vector_type(8)))  float    v8f;
typedef __attribute__((ext_vector_type(4)))  float    v4f;

union Frag16 { uint4 q[2]; v16h h; };

// ---------------------------------------------------------------------------
// f32 -> f16 conversion (embedding gather + linear weight)
// ---------------------------------------------------------------------------
__global__ void __launch_bounds__(256)
cvt_emb_kernel(const float* __restrict__ emb, const int* __restrict__ idx,
               _Float16* __restrict__ e16) {
    int i = blockIdx.x * blockDim.x + threadIdx.x;      // one float4 per thread
    int row = i >> 7;                                    // 128 float4 per row
    int c4  = i & 127;
    const float4 v = ((const float4*)(emb + (int64_t)idx[row] * DIM))[c4];
    v4h h;
    h[0] = (_Float16)v.x; h[1] = (_Float16)v.y;
    h[2] = (_Float16)v.z; h[3] = (_Float16)v.w;
    *(v4h*)(e16 + (int64_t)i * 4) = h;
}

__global__ void __launch_bounds__(256)
cvt_w_kernel(const float* __restrict__ w, _Float16* __restrict__ w16) {
    int i = blockIdx.x * blockDim.x + threadIdx.x;      // one float4 per thread
    const float4 v = ((const float4*)w)[i];
    v4h h;
    h[0] = (_Float16)v.x; h[1] = (_Float16)v.y;
    h[2] = (_Float16)v.z; h[3] = (_Float16)v.w;
    *(v4h*)(w16 + (int64_t)i * 4) = h;
}

// ---------------------------------------------------------------------------
// t = tanh(alpha * (E @ W^T + b)) -> f16, one 16x16 tile per wave, K = 512
// A(m,k) = E16[row][k], B(n,k) = W16[col][k]   (both k-contiguous)
// ---------------------------------------------------------------------------
__global__ void __launch_bounds__(256)
linear_wmma_kernel(const _Float16* __restrict__ E16,
                   const _Float16* __restrict__ W16,
                   const float*    __restrict__ bias,
                   _Float16*       __restrict__ T16) {
    const int wave = threadIdx.x >> 5;
    const int lane = threadIdx.x & 31;
    const int tile = blockIdx.x * 8 + wave;      // 1024 * 32 = 32768 tiles
    const int tr   = tile >> 5;                  // row tile  (NNODES/16)
    const int tc   = tile & 31;                  // col tile  (DIM/16)
    const int half = lane >> 4;
    const int row  = tr * 16 + (lane & 15);
    const int col  = tc * 16 + (lane & 15);

    v8f c = {};
    #pragma unroll 4
    for (int k0 = 0; k0 < DIM; k0 += 32) {
        Frag16 a, b;
        const _Float16* ap = E16 + (int64_t)row * DIM + k0 + half * 8;
        a.q[0] = *(const uint4*)ap;
        a.q[1] = *(const uint4*)(ap + 16);
        const _Float16* bp = W16 + (int64_t)col * DIM + k0 + half * 8;
        b.q[0] = *(const uint4*)bp;
        b.q[1] = *(const uint4*)(bp + 16);
        c = __builtin_amdgcn_wmma_f32_16x16x32_f16(false, a.h, false, b.h,
                                                   (short)0, c, false, false);
    }
    const float bv = bias[col];
    #pragma unroll
    for (int v = 0; v < 8; ++v) {
        const int orow = tr * 16 + v + half * 8;      // C/D layout: M = v + 8*half
        const float t  = tanhf(ALPHA * (c[v] + bv));
        T16[(int64_t)orow * DIM + col] = (_Float16)t;
    }
}

// ---------------------------------------------------------------------------
// Zero the 1 GB output with non-temporal stores (don't pollute L2 holding t16)
// ---------------------------------------------------------------------------
__global__ void __launch_bounds__(256)
zero_out_kernel(float* __restrict__ out, int n4) {
    v4f z = {0.f, 0.f, 0.f, 0.f};
    v4f* o = (v4f*)out;
    for (int i = blockIdx.x * blockDim.x + threadIdx.x; i < n4;
         i += gridDim.x * blockDim.x) {
        __builtin_nontemporal_store(z, o + i);
    }
}

// ---------------------------------------------------------------------------
// Fused sim = t @ t^T  +  per-row top-10  +  scatter into H^T.
// One wave owns 16 rows; A tile (16x512 f16 = contiguous 16KB) staged in LDS;
// loops over 1024 column tiles; t16 (16 MB) is L2-resident.
// ---------------------------------------------------------------------------
__global__ void __launch_bounds__(64)
topk_scatter_kernel(const _Float16* __restrict__ T16, float* __restrict__ out) {
    __shared__ _Float16 ldsA[2][16 * DIM];   // 2 waves * 16 KB
    __shared__ float    ldsC[2][16][16];     // 2 waves * 1 KB

    const int wave    = threadIdx.x >> 5;
    const int lane    = threadIdx.x & 31;
    const int half    = lane >> 4;
    const int rowTile = blockIdx.x * 2 + wave;       // 0..1023
    const int r0      = rowTile * 16;

    // stage A tile: rows r0..r0+15 are one contiguous 16 KB block
    {
        const uint4* src = (const uint4*)(T16 + (int64_t)r0 * DIM);
        uint4*       dst = (uint4*)ldsA[wave];
        #pragma unroll
        for (int i = lane; i < (16 * DIM) / 8; i += 32) dst[i] = src[i];
    }
    __syncthreads();

    float tv[TOPK];
    int   ti[TOPK];
    #pragma unroll
    for (int k = 0; k < TOPK; ++k) { tv[k] = -3.0e38f; ti[k] = 0; }

    for (int ct = 0; ct < NNODES / 16; ++ct) {
        const int col = ct * 16 + (lane & 15);
        v8f c = {};
        #pragma unroll 4
        for (int k0 = 0; k0 < DIM; k0 += 32) {
            Frag16 a, b;
            const _Float16* ap = ldsA[wave] + (lane & 15) * DIM + k0 + half * 8;
            a.q[0] = *(const uint4*)ap;                 // ds_load_b128
            a.q[1] = *(const uint4*)(ap + 16);
            const _Float16* bp = T16 + (int64_t)col * DIM + k0 + half * 8;
            b.q[0] = *(const uint4*)bp;                 // global_load_b128 (L2 hit)
            b.q[1] = *(const uint4*)(bp + 16);
            c = __builtin_amdgcn_wmma_f32_16x16x32_f16(false, a.h, false, b.h,
                                                       (short)0, c, false, false);
        }
        // dump 16x16 f32 tile: element (M = v + 8*half, N = lane&15)
        #pragma unroll
        for (int v = 0; v < 8; ++v)
            ldsC[wave][v + half * 8][lane & 15] = c[v];
        __syncthreads();

        if (lane < 16) {
            #pragma unroll
            for (int n = 0; n < 16; ++n) {
                const float val = ldsC[wave][lane][n];
                if (val > tv[TOPK - 1]) {
                    tv[TOPK - 1] = val;
                    ti[TOPK - 1] = ct * 16 + n;
                    #pragma unroll
                    for (int j = TOPK - 1; j > 0; --j) {   // static-index bubble-up
                        if (tv[j] > tv[j - 1]) {
                            float fv = tv[j]; tv[j] = tv[j - 1]; tv[j - 1] = fv;
                            int   iv = ti[j]; ti[j] = ti[j - 1]; ti[j - 1] = iv;
                        }
                    }
                }
            }
        }
        __syncthreads();
    }

    // scatter: out = H^T  =>  out[topidx * N + row] = 1
    if (lane < 16) {
        const int row = r0 + lane;
        #pragma unroll
        for (int k = 0; k < TOPK; ++k)
            out[(int64_t)ti[k] * NNODES + row] = 1.0f;
    }
}

// ---------------------------------------------------------------------------
extern "C" void kernel_launch(void* const* d_in, const int* in_sizes, int n_in,
                              void* d_out, int out_size, void* d_ws, size_t ws_size,
                              hipStream_t stream) {
    const int*   idx  = (const int*)d_in[0];
    const float* emb  = (const float*)d_in[1];
    const float* linw = (const float*)d_in[2];
    const float* linb = (const float*)d_in[3];
    float*       out  = (float*)d_out;

    char* ws = (char*)d_ws;
    _Float16* E16 = (_Float16*)ws;                                           // 16 MB
    _Float16* W16 = (_Float16*)(ws + (size_t)NNODES * DIM * 2);              // 512 KB
    _Float16* T16 = (_Float16*)(ws + (size_t)NNODES * DIM * 2
                                   + (size_t)DIM * DIM * 2);                 // 16 MB

    // 1) convert inputs to f16
    cvt_emb_kernel<<<(NNODES * DIM / 4) / 256, 256, 0, stream>>>(emb, idx, E16);
    cvt_w_kernel<<<(DIM * DIM / 4) / 256, 256, 0, stream>>>(linw, W16);

    // 2) t = tanh(alpha*(E @ W^T + b))  -> f16  (32768 tiles, 8 waves/block)
    linear_wmma_kernel<<<4096, 256, 0, stream>>>(E16, W16, linb, T16);

    // 3) zero 16384^2 f32 output (non-temporal)
    zero_out_kernel<<<8192, 256, 0, stream>>>(out, (NNODES / 4) * NNODES);

    // 4) fused sim GEMM + top-10 + scatter (1024 row tiles, 2 waves/block)
    topk_scatter_kernel<<<512, 64, 0, stream>>>(T16, out);
}
